// RLN_27453430956862
// MI455X (gfx1250) — matrix operand order repaired and checked by reading, not compiled, not run on hardware
//
#include <hip/hip_runtime.h>

#define NB   64
#define NC   124
#define NT   64
#define ND   16
#define NH   8
#define NLH  64
#define NSEQ 248
#define NG   256   // 4*LH
#define N1T  38    // N-tiles for W1 (600 -> 608)

typedef __attribute__((ext_vector_type(16))) _Float16 v16h;
typedef __attribute__((ext_vector_type(8)))  _Float16 v8h;
typedef __attribute__((ext_vector_type(8)))  float    v8f;

union frag_u { v16h v; v8h h[2]; };

__device__ __forceinline__ int lane_id() { return (int)(threadIdx.x & 31); }

__device__ __forceinline__ v16h make_frag(v8h lo, v8h hi) {
  frag_u u; u.h[0] = lo; u.h[1] = hi; return u.v;
}

__device__ __forceinline__ v8f wmma_f16(v16h a, v16h b, v8f c) {
  return __builtin_amdgcn_wmma_f32_16x16x32_f16(
      false, a, false, b, (short)0, c, false, false);
}

// A fragment 16x32 f16 from row-major [16][lda]; two contiguous 16B runs per lane.
// lanes 0-15: M=lane, K in {kb..kb+7, 16+kb..}, kb=0; lanes 16-31: kb=8.
__device__ __forceinline__ v16h load_a16(const _Float16* base, int lda) {
  int lane = lane_id();
  const _Float16* row = base + (lane & 15) * lda + ((lane < 16) ? 0 : 8);
  return make_frag(*(const v8h*)row, *(const v8h*)(row + 16));
}

// A fragment with only K<16 valid (K=16..31 zero)
__device__ __forceinline__ v16h load_a16_k16(const _Float16* base, int lda) {
  int lane = lane_id();
  const _Float16* row = base + (lane & 15) * lda + ((lane < 16) ? 0 : 8);
  v8h z = {};
  return make_frag(*(const v8h*)row, z);
}

// B fragment from pre-swizzled blob [32 lanes][16 halves]
__device__ __forceinline__ v16h load_bfrag(const _Float16* blob) {
  const _Float16* p = blob + lane_id() * 16;
  return make_frag(*(const v8h*)p, *(const v8h*)(p + 8));
}

// ---------------- Kernel 1: per-(b,c) encoder row ----------------
__global__ __launch_bounds__(32)
void enc_kernel(const float* __restrict__ x, const float* __restrict__ w_ch,
                const float* __restrict__ We, const float* __restrict__ be,
                const float* __restrict__ Wq, const float* __restrict__ bq,
                const float* __restrict__ Wk, const float* __restrict__ bk,
                const float* __restrict__ Wv, const float* __restrict__ bv,
                const float* __restrict__ Wo, const float* __restrict__ bo,
                const float* __restrict__ ln_g, const float* __restrict__ ln_b,
                const float* __restrict__ Wf, const float* __restrict__ bf,
                float* __restrict__ seq) {
  __shared__ float    s_xw[NT];
  __shared__ __align__(16) _Float16 s_h16[NT][ND];
  __shared__ float    s_h32[NT][ND];
  __shared__ __align__(16) _Float16 s_wfrag[4][32][16];  // Wq,Wk,Wv,Wo swizzled (K-pad zero)
  __shared__ float    s_q[NT][ND], s_k[NT][ND], s_v[NT][ND];
  __shared__ __align__(16) _Float16 s_o16[NT][ND];
  __shared__ float    s_We[ND], s_be[ND], s_lng[ND], s_lnb[ND];

  const int tid = threadIdx.x;
  const int bc  = blockIdx.x;
  const int b   = bc / NC, c = bc % NC;
  const float wch = w_ch[c];

  for (int t = tid; t < NT; t += 32) s_xw[t] = x[bc * NT + t] * wch;
  if (tid < ND) {
    s_We[tid] = We[tid]; s_be[tid] = be[tid];
    s_lng[tid] = ln_g[tid]; s_lnb[tid] = ln_b[tid];
  }
  // swizzle the 4 DxD weights into B-fragment blobs (zeros for elements j>=8)
  for (int i = tid; i < 4 * 32; i += 32) {
    int f = i >> 5, ln = i & 31;
    int nn = ln & 15;
    const float* W = (f == 0) ? Wq : (f == 1) ? Wk : (f == 2) ? Wv : Wo;
#pragma unroll
    for (int j = 0; j < 16; ++j)
      s_wfrag[f][ln][j] = (j < 8) ? (_Float16)W[((ln >> 4) * 8 + j) * ND + nn] : (_Float16)0.0f;
  }
  const int n  = tid & 15;
  const int mo = (tid < 16) ? 0 : 8;
  const float bqv = bq[n], bkv = bk[n], bvv = bv[n], bov = bo[n];
  __syncthreads();

  for (int i = tid; i < NT * ND; i += 32) {
    int t = i >> 4, d = i & 15;
    float hv = s_xw[t] * s_We[d] + s_be[d];
    s_h32[t][d] = hv;
    s_h16[t][d] = (_Float16)hv;
  }
  __syncthreads();

  const v16h wqf = load_bfrag(&s_wfrag[0][0][0]);
  const v16h wkf = load_bfrag(&s_wfrag[1][0][0]);
  const v16h wvf = load_bfrag(&s_wfrag[2][0][0]);
  const v16h wof = load_bfrag(&s_wfrag[3][0][0]);

  // Q/K/V projections: (64x16)@(16x16), 4 M-tiles, K padded 16->32
#pragma unroll
  for (int mt = 0; mt < 4; ++mt) {
    v16h a = load_a16_k16(&s_h16[mt * 16][0], ND);
    v8f accq = {}, acck = {}, accv = {};
    accq = wmma_f16(a, wqf, accq);
    acck = wmma_f16(a, wkf, acck);
    accv = wmma_f16(a, wvf, accv);
#pragma unroll
    for (int r = 0; r < 8; ++r) {
      int m = mt * 16 + r + mo;
      s_q[m][n] = accq[r] + bqv;
      s_k[m][n] = acck[r] + bkv;
      s_v[m][n] = accv[r] + bvv;
    }
  }
  __syncthreads();

  // Attention, HD=2: online softmax per (t, head) row in VALU
  for (int p = tid; p < NT * NH; p += 32) {
    int t = p >> 3, hh = p & 7;
    float2 qv = *(const float2*)&s_q[t][2 * hh];
    float mx = -3.0e38f, l = 0.f, o0 = 0.f, o1 = 0.f;
    for (int s = 0; s < NT; ++s) {
      float2 kv = *(const float2*)&s_k[s][2 * hh];
      float2 vv = *(const float2*)&s_v[s][2 * hh];
      float sc = (qv.x * kv.x + qv.y * kv.y) * 0.70710678f;
      float mn   = fmaxf(mx, sc);
      float corr = __expf(mx - mn);
      float e    = __expf(sc - mn);
      l  = l  * corr + e;
      o0 = o0 * corr + e * vv.x;
      o1 = o1 * corr + e * vv.y;
      mx = mn;
    }
    float inv = 1.0f / l;
    s_o16[t][2 * hh]     = (_Float16)(o0 * inv);
    s_o16[t][2 * hh + 1] = (_Float16)(o1 * inv);
  }
  __syncthreads();

  // Output projection + residual
#pragma unroll
  for (int mt = 0; mt < 4; ++mt) {
    v16h a = load_a16_k16(&s_o16[mt * 16][0], ND);
    v8f acc = {};
    acc = wmma_f16(a, wof, acc);
#pragma unroll
    for (int r = 0; r < 8; ++r) {
      int m = mt * 16 + r + mo;
      s_h32[m][n] += acc[r] + bov;
    }
  }
  __syncthreads();

  // LayerNorm per row (D=16)
  for (int t = tid; t < NT; t += 32) {
    float mu = 0.f;
    for (int d = 0; d < ND; ++d) mu += s_h32[t][d];
    mu *= (1.0f / ND);
    float var = 0.f;
    for (int d = 0; d < ND; ++d) { float df = s_h32[t][d] - mu; var += df * df; }
    var *= (1.0f / ND);
    float rs = rsqrtf(var + 1e-5f);
    for (int d = 0; d < ND; ++d)
      s_h32[t][d] = (s_h32[t][d] - mu) * rs * s_lng[d] + s_lnb[d];
  }
  __syncthreads();

  // x1 (mean over T) -> seq[b][c][:]; x2 (xw@Wf) -> seq[b][C+c][:]
  if (tid < ND) {
    float s1 = 0.f;
    for (int t = 0; t < NT; ++t) s1 += s_h32[t][tid];
    seq[(b * NSEQ + c) * ND + tid] = s1 * (1.0f / NT);
    float a2 = bf[tid];
    for (int t = 0; t < NT; ++t) a2 += s_xw[t] * Wf[t * ND + tid];
    seq[(b * NSEQ + NC + c) * ND + tid] = (a2 > 0.f) ? a2 : 0.01f * a2;
  }
}

// -------- Pack kernel: seq -> x A-fragments (f16), W1 -> B-fragments (f16) --------
__global__ __launch_bounds__(256)
void pack_kernel(const float* __restrict__ seq, const float* __restrict__ W1,
                 _Float16* __restrict__ xfrag, _Float16* __restrict__ w1frag) {
  const int gid = blockIdx.x * blockDim.x + threadIdx.x;
  const int stride = gridDim.x * blockDim.x;

  // xfrag: [s][mt][lane][8] low-half A fragments (K<16; high half implied zero)
  const int NXA = NSEQ * 4 * 32;
  for (int i = gid; i < NXA; i += stride) {
    int lane = i & 31, mt = (i >> 5) & 3, s = i >> 7;
    int m  = lane & 15;
    int kb = (lane < 16) ? 0 : 8;
    const float* src = &seq[((size_t)(mt * 16 + m) * NSEQ + s) * ND + kb];
    _Float16* dst = &xfrag[(size_t)i * 8];
#pragma unroll
    for (int j = 0; j < 8; ++j) dst[j] = (_Float16)src[j];
  }

  // w1frag: [nt][kt][lane][16] B fragments, zero-padded K>=248 / N>=600
  const int NXB = N1T * 8 * 32;
  for (int i = gid; i < NXB; i += stride) {
    int lane = i & 31, kt = (i >> 5) & 7, nt = i >> 8;
    int nn = lane & 15;
    int kb = (lane < 16) ? 0 : 8;
    int ng = nt * 16 + nn;
    _Float16* dst = &w1frag[(size_t)i * 16];
#pragma unroll
    for (int j = 0; j < 16; ++j) {
      int k = kt * 32 + kb + j + ((j >= 8) ? 8 : 0);
      dst[j] = (k < NSEQ && ng < 600) ? (_Float16)W1[k * 600 + ng] : (_Float16)0.0f;
    }
  }
}

// ---------------- Kernel 2: persistent LSTM (1 workgroup, 16 waves) ----------------
__global__ __launch_bounds__(512)
void lstm_kernel(const _Float16* __restrict__ xfrag,
                 const float* __restrict__ W_ih, const float* __restrict__ W_hh,
                 const float* __restrict__ b_ih, const float* __restrict__ b_hh,
                 float* __restrict__ feat) {
  __shared__ __align__(32) _Float16 s_wfrag[48][32][16]; // 0..15 W_ih, 16..31 hh slab0, 32..47 hh slab1
  __shared__ __align__(16) _Float16 s_h[NB][NLH];

  const int tid = threadIdx.x;
  // swizzle weights into B-fragment blobs
  for (int i = tid; i < 48 * 32; i += 512) {
    int f = i >> 5, ln = i & 31;
    int nn = ln & 15;
    int kb = (ln < 16) ? 0 : 8;
    _Float16* dst = &s_wfrag[f][ln][0];
    if (f < 16) {
      int col = f * 16 + nn;
#pragma unroll
      for (int j = 0; j < 16; ++j)
        dst[j] = (j < 8) ? (_Float16)W_ih[(kb + j) * NG + col] : (_Float16)0.0f;
    } else {
      int slab = (f - 16) >> 4;
      int col  = ((f - 16) & 15) * 16 + nn;
#pragma unroll
      for (int j = 0; j < 16; ++j)
        dst[j] = (_Float16)W_hh[(slab * 32 + kb + j + ((j >= 8) ? 8 : 0)) * NG + col];
    }
  }
  for (int i = tid; i < NB * NLH; i += 512)
    s_h[i >> 6][i & 63] = (_Float16)0.0f;

  const int wave = tid >> 5;
  const int mt   = wave >> 2;               // batch tile
  const int ntq  = wave & 3;                // gate-quadruple column group
  const int n    = tid & 15;
  const int mo   = ((tid & 31) < 16) ? 0 : 8;
  const int u    = ntq * 16 + n;            // hidden unit 0..63
  const int nti = ntq, ntf = 4 + ntq, ntg = 8 + ntq, nto = 12 + ntq;
  const bool featlane = (u == NLH - 1);

  const float biasI = b_ih[nti * 16 + n] + b_hh[nti * 16 + n];
  const float biasF = b_ih[ntf * 16 + n] + b_hh[ntf * 16 + n];
  const float biasG = b_ih[ntg * 16 + n] + b_hh[ntg * 16 + n];
  const float biasO = b_ih[nto * 16 + n] + b_hh[nto * 16 + n];

  __syncthreads();

  // hoist all 12 weight fragments into registers (loop-invariant)
  const v16h wI0 = load_bfrag(&s_wfrag[nti][0][0]);
  const v16h wI1 = load_bfrag(&s_wfrag[16 + nti][0][0]);
  const v16h wI2 = load_bfrag(&s_wfrag[32 + nti][0][0]);
  const v16h wF0 = load_bfrag(&s_wfrag[ntf][0][0]);
  const v16h wF1 = load_bfrag(&s_wfrag[16 + ntf][0][0]);
  const v16h wF2 = load_bfrag(&s_wfrag[32 + ntf][0][0]);
  const v16h wG0 = load_bfrag(&s_wfrag[ntg][0][0]);
  const v16h wG1 = load_bfrag(&s_wfrag[16 + ntg][0][0]);
  const v16h wG2 = load_bfrag(&s_wfrag[32 + ntg][0][0]);
  const v16h wO0 = load_bfrag(&s_wfrag[nto][0][0]);
  const v16h wO1 = load_bfrag(&s_wfrag[16 + nto][0][0]);
  const v16h wO2 = load_bfrag(&s_wfrag[32 + nto][0][0]);

  float cReg[8];
#pragma unroll
  for (int r = 0; r < 8; ++r) cReg[r] = 0.f;

  const int lane = tid & 31;
  for (int s = 0; s < NSEQ; ++s) {
    v8h z = {};
    v8h axlo = *(const v8h*)(xfrag + ((size_t)(s * 4 + mt) * 32 + lane) * 8);
    v16h ax  = make_frag(axlo, z);
    v16h ah0 = load_a16(&s_h[mt * 16][0],  NLH);
    v16h ah1 = load_a16(&s_h[mt * 16][32], NLH);

    v8f gi = {}, gf = {}, gg = {}, go = {};
    gi = wmma_f16(ax, wI0, gi); gi = wmma_f16(ah0, wI1, gi); gi = wmma_f16(ah1, wI2, gi);
    gf = wmma_f16(ax, wF0, gf); gf = wmma_f16(ah0, wF1, gf); gf = wmma_f16(ah1, wF2, gf);
    gg = wmma_f16(ax, wG0, gg); gg = wmma_f16(ah0, wG1, gg); gg = wmma_f16(ah1, wG2, gg);
    go = wmma_f16(ax, wO0, go); go = wmma_f16(ah0, wO1, go); go = wmma_f16(ah1, wO2, go);

    float hN[8];
#pragma unroll
    for (int r = 0; r < 8; ++r) {
      float si = 1.f / (1.f + __expf(-(gi[r] + biasI)));
      float sf = 1.f / (1.f + __expf(-(gf[r] + biasF)));
      float so = 1.f / (1.f + __expf(-(go[r] + biasO)));
      cReg[r] = sf * cReg[r] + si * tanhf(gg[r] + biasG);
      hN[r]   = so * tanhf(cReg[r]);
    }
    __syncthreads();   // all reads of old h complete
#pragma unroll
    for (int r = 0; r < 8; ++r)
      s_h[mt * 16 + r + mo][u] = (_Float16)hN[r];
    if (featlane) {
#pragma unroll
      for (int r = 0; r < 8; ++r)
        feat[(size_t)(mt * 16 + r + mo) * NSEQ + s] = hN[r];
    }
    __syncthreads();   // new h visible before next step
  }
}

// ---------------- Kernel 3: head — feat@W1 (WMMA) then tiny W2 layer ----------------
__global__ __launch_bounds__(512)
void head_kernel(const float* __restrict__ feat, const _Float16* __restrict__ w1frag,
                 const float* __restrict__ b1, const float* __restrict__ W2,
                 const float* __restrict__ b2, float* __restrict__ out) {
  __shared__ __align__(16) _Float16 s_f16[NB][256];   // K padded 248->256
  __shared__ float s_w2[3600];
  __shared__ float s_out[NB][8];

  const int tid = threadIdx.x;
  for (int i = tid; i < NB * 256; i += 512) {
    int bb = i >> 8, k = i & 255;
    s_f16[bb][k] = (k < NSEQ) ? (_Float16)feat[bb * NSEQ + k] : (_Float16)0.0f;
  }
  for (int i = tid; i < 3600; i += 512) s_w2[i] = W2[i];
  for (int i = tid; i < NB * 8; i += 512) s_out[i >> 3][i & 7] = 0.0f;
  __syncthreads();

  const int lane = tid & 31;
  const int n    = lane & 15;
  const int mo   = (lane < 16) ? 0 : 8;
  const int wave = tid >> 5;

  for (int tile = wave; tile < 4 * N1T; tile += 16) {
    int mt = tile & 3;
    int nt = tile >> 2;
    int ng = nt * 16 + n;
    v8f acc = {};
#pragma unroll
    for (int kt = 0; kt < 8; ++kt) {
      v16h a  = load_a16(&s_f16[mt * 16][kt * 32], 256);
      v16h bw = load_bfrag(w1frag + (size_t)(nt * 8 + kt) * 32 * 16);
      acc = wmma_f16(a, bw, acc);
    }
    if (ng < 600) {
      float bias = b1[ng];
#pragma unroll
      for (int r = 0; r < 8; ++r) {
        int m = mt * 16 + r + mo;
        float yv = acc[r] + bias;
        yv = (yv > 0.f) ? yv : 0.01f * yv;      // leaky on layer-1 output
#pragma unroll
        for (int j = 0; j < 6; ++j)
          atomicAdd(&s_out[m][j], yv * s_w2[ng * 6 + j]);
      }
    }
  }
  __syncthreads();
  if (tid < NB * 6) {
    int bb = tid / 6, j = tid % 6;
    float v = s_out[bb][j] + b2[j];
    out[tid] = (v > 0.f) ? v : 0.01f * v;
  }
}

extern "C" void kernel_launch(void* const* d_in, const int* in_sizes, int n_in,
                              void* d_out, int out_size, void* d_ws, size_t ws_size,
                              hipStream_t stream) {
  const float* x    = (const float*)d_in[0];
  const float* w_ch = (const float*)d_in[1];
  const float* We   = (const float*)d_in[2];
  const float* be   = (const float*)d_in[3];
  const float* Wq   = (const float*)d_in[4];
  const float* bq   = (const float*)d_in[5];
  const float* Wk   = (const float*)d_in[6];
  const float* bk   = (const float*)d_in[7];
  const float* Wv   = (const float*)d_in[8];
  const float* bv   = (const float*)d_in[9];
  const float* Wo   = (const float*)d_in[10];
  const float* bo   = (const float*)d_in[11];
  const float* ln_g = (const float*)d_in[12];
  const float* ln_b = (const float*)d_in[13];
  const float* Wf   = (const float*)d_in[14];
  const float* bf   = (const float*)d_in[15];
  const float* W_ih = (const float*)d_in[16];
  const float* W_hh = (const float*)d_in[17];
  const float* b_ih = (const float*)d_in[18];
  const float* b_hh = (const float*)d_in[19];
  const float* W1   = (const float*)d_in[20];
  const float* b1   = (const float*)d_in[21];
  const float* W2   = (const float*)d_in[22];
  const float* b2   = (const float*)d_in[23];

  // workspace layout (16/32-byte aligned sections)
  float* seq_ws  = (float*)d_ws;                         // B*SEQ*D f32
  float* feat_ws = seq_ws + (size_t)NB * NSEQ * ND;      // B*SEQ   f32
  _Float16* xfrag_ws  = (_Float16*)(feat_ws + (size_t)NB * NSEQ);   // SEQ*4*32*8 f16
  _Float16* w1frag_ws = xfrag_ws + (size_t)NSEQ * 4 * 32 * 8;       // 38*8*32*16 f16

  enc_kernel<<<dim3(NB * NC), dim3(32), 0, stream>>>(
      x, w_ch, We, be, Wq, bq, Wk, bk, Wv, bv, Wo, bo, ln_g, ln_b, Wf, bf, seq_ws);
  pack_kernel<<<dim3(64), dim3(256), 0, stream>>>(seq_ws, W1, xfrag_ws, w1frag_ws);
  lstm_kernel<<<dim3(1), dim3(512), 0, stream>>>(
      xfrag_ws, W_ih, W_hh, b_ih, b_hh, feat_ws);
  head_kernel<<<dim3(1), dim3(512), 0, stream>>>(
      feat_ws, w1frag_ws, b1, W2, b2, (float*)d_out);
}
